// MAMBA2D_60232621359562
// MI455X (gfx1250) — compile-verified, matrix operand order checked by reading, not run on hardware
//
#include <hip/hip_runtime.h>
#include <hip/hip_bf16.h>
#include <math.h>

// ---------------- types / WMMA helpers (gfx1250 wave32) ----------------
typedef __attribute__((ext_vector_type(16))) __bf16 v16bf;
typedef __attribute__((ext_vector_type(8)))  __bf16 v8bf;
typedef __attribute__((ext_vector_type(8)))  float  v8f;

#define DEV static __device__ __forceinline__

DEV v8f wmma_bf(v16bf a, v16bf b, v8f c) {
  // D = A(16x32) * B(32x16) + C  (f32 accum)
  return __builtin_amdgcn_wmma_f32_16x16x32_bf16(false, a, false, b, (short)0, c, false, false);
}

// Fragment loader for 16x32 bf16 A-tile (row-major, ld elements) or the mirrored
// B-tile (column n per lane, B stored row-major [N,K]).
// Lane L<16: row r0+L, K = {k0..k0+7, k0+16..k0+23}
// Lane L>=16: row r0+L-16, K = {k0+8..k0+15, k0+24..k0+31}
DEV v16bf load_frag(const __bf16* base, int ld, int r0, int k0) {
  const int lane = (int)(threadIdx.x & 31u);
  const int idx  = lane & 15;
  const int half = lane >> 4;
  const __bf16* p = base + (size_t)(r0 + idx) * (size_t)ld + (size_t)(k0 + half * 8);
  v8bf lo = *(const v8bf*)(p);
  v8bf hi = *(const v8bf*)(p + 16);
  v16bf r;
#pragma unroll
  for (int i = 0; i < 8; ++i) { r[i] = lo[i]; r[i + 8] = hi[i]; }
  return r;
}

DEV float sigmoidf_(float x) { return 1.0f / (1.0f + __expf(-x)); }
DEV float softplusf_(float x) { return (x > 20.f) ? x : log1pf(__expf(x)); }

// ---------------- problem constants ----------------
#define ROWS   8192      // 128 seq * 64 tokens
#define NSEQ   128
#define LSEQ   64
#define DMODEL 512
#define DINNER 1024
#define NHEADS 16
#define HEADD  64
#define CONVD  1280
#define NPROJ  2320
#define NPAD   2368      // padded to multiple of 64 for GEMM tiles
#define EPSV   1e-5f

// ---------------- elementwise prep kernels ----------------
__global__ void k_cvt_bf(const float* __restrict__ s, __bf16* __restrict__ d, int n) {
  int i = blockIdx.x * 256 + threadIdx.x;
  if (i < n) d[i] = (__bf16)s[i];
}

__global__ void k_prep_inw(const float* __restrict__ w, __bf16* __restrict__ wb) {
  int i = blockIdx.x * 256 + threadIdx.x;          // over 2368*512
  int r = i >> 9;
  wb[i] = (r < NPROJ) ? (__bf16)w[i] : (__bf16)0.0f;
}

// fold fc_w (512,2048) -> (512,1024) using v2==v1, h2==h1 duplication
__global__ void k_prep_fin(const float* __restrict__ fc, __bf16* __restrict__ wf) {
  int i = blockIdx.x * 256 + threadIdx.x;          // over 512*1024
  int o = i >> 10, c = i & 1023;
  float v;
  if (c < 512) v = fc[o * 2048 + c] + fc[o * 2048 + c + 512];
  else { int cc = c - 512; v = fc[o * 2048 + 1024 + cc] + fc[o * 2048 + 1536 + cc]; }
  wf[i] = (__bf16)v;
}

__global__ void k_gather_h(const float* __restrict__ x, __bf16* __restrict__ u) {
  int i = blockIdx.x * 256 + threadIdx.x;          // over 8192*512
  u[i] = (__bf16)x[i];
}

// vertical sequences: row r=(b*64+w)*64+t  <- x[b,t,w,:]
__global__ void k_gather_v(const float* __restrict__ x, __bf16* __restrict__ u) {
  int i = blockIdx.x * 256 + threadIdx.x;
  int c = i & 511, row = i >> 9;
  int b = row >> 12, w = (row >> 6) & 63, t = row & 63;
  size_t src = ((size_t)((b * 64 + t) * 64 + w) << 9) + c;
  u[i] = (__bf16)x[src];
}

// ---------------- generic bf16 GEMM: C[M,N] = A[M,K] * B[N,K]^T (+bias) ----------------
// block: 256 threads (8 waves), tile 128(M) x 64(N); wave = 4 m-subtiles x 1 n-subtile.
// Unroll-by-2 ping-pong: two fixed fragment register sets, loads of one set overlap
// the WMMAs of the other; K must be a multiple of 64 (all call sites satisfy this).
__global__ void k_gemm(const __bf16* __restrict__ A, const __bf16* __restrict__ B,
                       const float* __restrict__ bias, float* __restrict__ C,
                       int M, int N, int K, int ldc) {
  const int wid  = (int)(threadIdx.x >> 5);
  const int lane = (int)(threadIdx.x & 31u);
  const int idx  = lane & 15, half = lane >> 4;
  const int m0 = blockIdx.y * 128 + (wid >> 2) * 64;
  const int n0 = blockIdx.x * 64 + (wid & 3) * 16;
  (void)M; (void)N;
  v8f acc[4] = {};

  // set 0 preload (k = 0)
  v16bf b0 = load_frag(B, K, n0, 0);
  v16bf a0[4];
#pragma unroll
  for (int i = 0; i < 4; ++i) a0[i] = load_frag(A, K, m0 + i * 16, 0);

  for (int k = 0; k < K; k += 64) {
    // load set 1 (k+32) while computing set 0
    v16bf b1 = load_frag(B, K, n0, k + 32);
    v16bf a1[4];
#pragma unroll
    for (int i = 0; i < 4; ++i) a1[i] = load_frag(A, K, m0 + i * 16, k + 32);
    if (k + 64 < K) {
      __builtin_prefetch(B + (size_t)(n0 + idx) * K + k + 64, 0, 1);
      __builtin_prefetch(A + (size_t)(m0 + idx) * K + k + 64, 0, 1);
    }
#pragma unroll
    for (int i = 0; i < 4; ++i) acc[i] = wmma_bf(a0[i], b0, acc[i]);

    // load set 0 (k+64) while computing set 1
    if (k + 64 < K) {
      b0 = load_frag(B, K, n0, k + 64);
#pragma unroll
      for (int i = 0; i < 4; ++i) a0[i] = load_frag(A, K, m0 + i * 16, k + 64);
    }
#pragma unroll
    for (int i = 0; i < 4; ++i) acc[i] = wmma_bf(a1[i], b1, acc[i]);
  }

  const float bv = bias ? bias[n0 + idx] : 0.0f;
#pragma unroll
  for (int i = 0; i < 4; ++i) {
#pragma unroll
    for (int v = 0; v < 8; ++v) {
      int row = m0 + i * 16 + v + 8 * half;
      C[(size_t)row * ldc + n0 + idx] = acc[i][v] + bv;
    }
  }
}

// ---------------- depthwise causal conv (width 4) + bias + silu -> bf16 ----------------
__global__ void k_conv_silu(const float* __restrict__ zx, const float* __restrict__ cw,
                            const float* __restrict__ cb, __bf16* __restrict__ xbc) {
  int c = blockIdx.x * 256 + threadIdx.x;          // 0..1279
  int s = blockIdx.y;                              // sequence
  if (c >= CONVD) return;
  float w0 = cw[c * 4 + 0], w1 = cw[c * 4 + 1], w2 = cw[c * 4 + 2], w3 = cw[c * 4 + 3];
  float b = cb[c];
  float h0 = 0.f, h1 = 0.f, h2 = 0.f;
  for (int t = 0; t < LSEQ; ++t) {
    float xt = zx[(size_t)(s * LSEQ + t) * NPAD + DINNER + c];
    float a = h0 * w0 + h1 * w1 + h2 * w2 + xt * w3 + b;
    float sv = a * sigmoidf_(a);
    xbc[(size_t)(s * LSEQ + t) * CONVD + c] = (__bf16)sv;
    h0 = h1; h1 = h2; h2 = xt;
  }
}

// ---------------- dt = softplus(dt+bias); dtA = -exp(A_log)*dt; cumsum over t ----------------
__global__ void k_dt_cs(const float* __restrict__ zx, const float* __restrict__ A_log,
                        const float* __restrict__ dt_bias,
                        float* __restrict__ dt_o, float* __restrict__ cs_o) {
  int s = blockIdx.x;
  int h = threadIdx.x;
  if (h >= NHEADS) return;
  float nA = -__expf(A_log[h]);
  float db = dt_bias[h];
  float run = 0.f;
  for (int t = 0; t < LSEQ; ++t) {
    float raw = zx[(size_t)(s * LSEQ + t) * NPAD + (DINNER + CONVD) + h] + db;
    float dtv = softplusf_(raw);
    run += dtv * nA;
    dt_o[(size_t)(s * NHEADS + h) * LSEQ + t] = dtv;
    cs_o[(size_t)(s * NHEADS + h) * LSEQ + t] = run;
  }
}

// ---------------- per-(seq,head): G=C·Bᵀ, M=G∘L, Y=M·(dt⊙x); y+=D*x ----------------
__global__ void k_intra(const __bf16* __restrict__ xbc, const float* __restrict__ dt_b,
                        const float* __restrict__ cs_b, const float* __restrict__ Dp,
                        float* __restrict__ yraw) {
  __shared__ float cs_s[LSEQ];
  __shared__ float dt_s[LSEQ];
  __shared__ __bf16 Mlds[LSEQ * LSEQ];   // [t][s]
  __shared__ __bf16 XT[LSEQ * LSEQ];     // [p][s] = dt[s]*x[s][p]
  const int seq = blockIdx.x, head = blockIdx.y;
  const int tid = (int)threadIdx.x;
  const int wid = tid >> 5, lane = tid & 31, idx = lane & 15, half = lane >> 4;

  if (tid < LSEQ) {
    cs_s[tid] = cs_b[(size_t)(seq * NHEADS + head) * LSEQ + tid];
    dt_s[tid] = dt_b[(size_t)(seq * NHEADS + head) * LSEQ + tid];
  }
  __syncthreads();

  // build XT = (dt ⊙ x)^T in LDS
  for (int i = tid; i < LSEQ * LSEQ; i += 256) {
    int s = i & 63, p = i >> 6;
    float xv = (float)xbc[(size_t)(seq * LSEQ + s) * CONVD + head * HEADD + p];
    XT[p * LSEQ + s] = (__bf16)(dt_s[s] * xv);
  }

  // G = Cm(64x128) * Bm(64x128)^T via WMMA; 16 tiles, 2 per wave
  const __bf16* Cbase = xbc + (size_t)(seq * LSEQ) * CONVD + (DINNER + 128); // C channels
  const __bf16* Bbase = xbc + (size_t)(seq * LSEQ) * CONVD + DINNER;        // B channels
  v8f accG[2] = {};
#pragma unroll
  for (int u = 0; u < 2; ++u) {
    int tile = wid * 2 + u, ti = tile >> 2, si = tile & 3;
#pragma unroll
    for (int k = 0; k < 128; k += 32) {
      v16bf a = load_frag(Cbase, CONVD, ti * 16, k);
      v16bf b = load_frag(Bbase, CONVD, si * 16, k);
      accG[u] = wmma_bf(a, b, accG[u]);
    }
  }
  // apply decay mask L and stash M in LDS
#pragma unroll
  for (int u = 0; u < 2; ++u) {
    int tile = wid * 2 + u, ti = tile >> 2, si = tile & 3;
#pragma unroll
    for (int v = 0; v < 8; ++v) {
      int t = ti * 16 + v + 8 * half;
      int s = si * 16 + idx;
      float m = (t >= s) ? accG[u][v] * __expf(cs_s[t] - cs_s[s]) : 0.0f;
      Mlds[t * LSEQ + s] = (__bf16)m;
    }
  }
  __syncthreads();

  // Y = M(64x64) * Xd(64x64), fragments from LDS (ds_load path)
  v8f accY[2] = {};
#pragma unroll
  for (int u = 0; u < 2; ++u) {
    int tile = wid * 2 + u, ti = tile >> 2, pi = tile & 3;
#pragma unroll
    for (int k = 0; k < 64; k += 32) {
      v16bf a = load_frag(Mlds, LSEQ, ti * 16, k);
      v16bf b = load_frag(XT,   LSEQ, pi * 16, k);
      accY[u] = wmma_bf(a, b, accY[u]);
    }
  }
  const float Dh = Dp[head];
#pragma unroll
  for (int u = 0; u < 2; ++u) {
    int tile = wid * 2 + u, ti = tile >> 2, pi = tile & 3;
#pragma unroll
    for (int v = 0; v < 8; ++v) {
      int t = ti * 16 + v + 8 * half;
      int p = pi * 16 + idx;
      float xv = (float)xbc[(size_t)(seq * LSEQ + t) * CONVD + head * HEADD + p];
      yraw[(size_t)(seq * LSEQ + t) * DINNER + head * HEADD + p] = accY[u][v] + Dh * xv;
    }
  }
}

// ---------------- y *= silu(z); RMS-norm(1024) * norm_w -> bf16 ----------------
__global__ void k_gate_norm(const float* __restrict__ yraw, const float* __restrict__ zx,
                            const float* __restrict__ nw, __bf16* __restrict__ yn) {
  __shared__ float red[256];
  const int row = blockIdx.x;
  const int tid = (int)threadIdx.x;
  float vals[4];
  float ss = 0.f;
#pragma unroll
  for (int j = 0; j < 4; ++j) {
    int c = tid + j * 256;
    float z = zx[(size_t)row * NPAD + c];
    float y = yraw[(size_t)row * DINNER + c] * (z * sigmoidf_(z));
    vals[j] = y;
    ss += y * y;
  }
  red[tid] = ss;
  __syncthreads();
  for (int off = 128; off > 0; off >>= 1) {
    if (tid < off) red[tid] += red[tid + off];
    __syncthreads();
  }
  float scale = rsqrtf(red[0] * (1.0f / (float)DINNER) + EPSV);
#pragma unroll
  for (int j = 0; j < 4; ++j) {
    int c = tid + j * 256;
    yn[(size_t)row * DINNER + c] = (__bf16)(vals[j] * scale * nw[c]);
  }
}

// ---------------- build final GEMM A: cols[0:512]=v1 gathered, cols[512:1024]=h1 ----------------
__global__ void k_build_afin(const float* __restrict__ v1, const float* __restrict__ h1,
                             __bf16* __restrict__ af) {
  int i = blockIdx.x * 256 + threadIdx.x;          // over 8192*1024
  int row = i >> 10, c = i & 1023;
  float v;
  if (c < 512) {
    int b = row >> 12, h = (row >> 6) & 63, w = row & 63;
    int vr = (b << 12) + (w << 6) + h;             // v1 row (b*64+w)*64+h
    v = v1[(size_t)vr * DMODEL + c];
  } else {
    v = h1[(size_t)row * DMODEL + (c - 512)];
  }
  af[i] = (__bf16)v;
}

// ---------------- host launch ----------------
extern "C" void kernel_launch(void* const* d_in, const int* in_sizes, int n_in,
                              void* d_out, int out_size, void* d_ws, size_t ws_size,
                              hipStream_t stream) {
  (void)in_sizes; (void)n_in; (void)out_size; (void)ws_size;
  const float* x      = (const float*)d_in[0];
  const float* p_in_w [2] = {(const float*)d_in[1],  (const float*)d_in[9]};
  const float* p_cw   [2] = {(const float*)d_in[2],  (const float*)d_in[10]};
  const float* p_cb   [2] = {(const float*)d_in[3],  (const float*)d_in[11]};
  const float* p_Alog [2] = {(const float*)d_in[4],  (const float*)d_in[12]};
  const float* p_dtb  [2] = {(const float*)d_in[5],  (const float*)d_in[13]};
  const float* p_D    [2] = {(const float*)d_in[6],  (const float*)d_in[14]};
  const float* p_nw   [2] = {(const float*)d_in[7],  (const float*)d_in[15]};
  const float* p_ow   [2] = {(const float*)d_in[8],  (const float*)d_in[16]};
  const float* fc_w   = (const float*)d_in[17];
  const float* fc_b   = (const float*)d_in[18];
  float* out = (float*)d_out;

  // workspace carve-up (256B aligned)
  char* ws = (char*)d_ws;
  size_t off = 0;
  auto take = [&](size_t bytes) { char* p = ws + off; off += (bytes + 255) & ~(size_t)255; return p; };
  __bf16* u_bf    = (__bf16*)take((size_t)ROWS * DMODEL * 2);
  __bf16* win[2]  = {(__bf16*)take((size_t)NPAD * DMODEL * 2), (__bf16*)take((size_t)NPAD * DMODEL * 2)};
  __bf16* wout[2] = {(__bf16*)take((size_t)DMODEL * DINNER * 2), (__bf16*)take((size_t)DMODEL * DINNER * 2)};
  __bf16* wfin    = (__bf16*)take((size_t)DMODEL * DINNER * 2);
  char*   zx_raw  = take((size_t)ROWS * NPAD * 4);
  float*  zx      = (float*)zx_raw;
  __bf16* xbc     = (__bf16*)take((size_t)ROWS * CONVD * 2);
  float*  dt_buf  = (float*)take((size_t)NSEQ * NHEADS * LSEQ * 4);
  float*  cs_buf  = (float*)take((size_t)NSEQ * NHEADS * LSEQ * 4);
  float*  yraw    = (float*)take((size_t)ROWS * DINNER * 4);
  __bf16* yn      = (__bf16*)take((size_t)ROWS * DINNER * 2);
  float*  dres[2] = {(float*)take((size_t)ROWS * DMODEL * 4), (float*)take((size_t)ROWS * DMODEL * 4)}; // h1, v1
  __bf16* afin    = (__bf16*)zx_raw;  // alias: zx dead before final GEMM

  // ---- weight prep ----
  k_prep_inw<<<dim3((NPAD * DMODEL) / 256), 256, 0, stream>>>(p_in_w[0], win[0]);
  k_prep_inw<<<dim3((NPAD * DMODEL) / 256), 256, 0, stream>>>(p_in_w[1], win[1]);
  k_cvt_bf  <<<dim3((DMODEL * DINNER) / 256), 256, 0, stream>>>(p_ow[0], wout[0], DMODEL * DINNER);
  k_cvt_bf  <<<dim3((DMODEL * DINNER) / 256), 256, 0, stream>>>(p_ow[1], wout[1], DMODEL * DINNER);
  k_prep_fin<<<dim3((DMODEL * DINNER) / 256), 256, 0, stream>>>(fc_w, wfin);

  // ---- two directions (flip is batch-dim only => h2==h1, v2==v1) ----
  for (int dir = 0; dir < 2; ++dir) {
    if (dir == 0) k_gather_h<<<dim3((ROWS * DMODEL) / 256), 256, 0, stream>>>(x, u_bf);
    else          k_gather_v<<<dim3((ROWS * DMODEL) / 256), 256, 0, stream>>>(x, u_bf);

    // in_proj: (8192,512) x (2368,512)^T -> (8192,2368)
    k_gemm<<<dim3(NPAD / 64, ROWS / 128), 256, 0, stream>>>(u_bf, win[dir], nullptr, zx,
                                                            ROWS, NPAD, DMODEL, NPAD);
    k_conv_silu<<<dim3((CONVD + 255) / 256, NSEQ), 256, 0, stream>>>(zx, p_cw[dir], p_cb[dir], xbc);
    k_dt_cs<<<dim3(NSEQ), 32, 0, stream>>>(zx, p_Alog[dir], p_dtb[dir], dt_buf, cs_buf);
    k_intra<<<dim3(NSEQ, NHEADS), 256, 0, stream>>>(xbc, dt_buf, cs_buf, p_D[dir], yraw);
    k_gate_norm<<<dim3(ROWS), 256, 0, stream>>>(yraw, zx, p_nw[dir], yn);
    // out_proj: (8192,1024) x (512,1024)^T -> (8192,512)
    k_gemm<<<dim3(DMODEL / 64, ROWS / 128), 256, 0, stream>>>(yn, wout[dir], nullptr, dres[dir],
                                                              ROWS, DMODEL, DINNER, DMODEL);
  }

  // ---- final fused FC: out = [v1_gathered | h1] @ wfin^T + fc_b ----
  k_build_afin<<<dim3((ROWS * DINNER) / 256), 256, 0, stream>>>(dres[1], dres[0], afin);
  k_gemm<<<dim3(DMODEL / 64, ROWS / 128), 256, 0, stream>>>(afin, wfin, fc_b, out,
                                                            ROWS, DMODEL, DINNER, DMODEL);
}
